// SelfAttention_54236847013919
// MI455X (gfx1250) — matrix operand order repaired and checked by reading, not compiled, not run on hardware
//
#include <hip/hip_runtime.h>

#define B_ 4
#define S_ 2048
#define D_ 1024
#define U_ 1024

typedef __attribute__((ext_vector_type(16))) __bf16 v16bf;
typedef __attribute__((ext_vector_type(8)))  float  v8f;

union Frag {
    v16bf v;
    uint4 q[2];
    unsigned short h[16];
};

static __device__ __forceinline__ unsigned short f2bf(float f) {
    union { float f; unsigned int u; } c; c.f = f;
    unsigned int u = c.u;
    unsigned int r = (u + 0x7FFFu + ((u >> 16) & 1u)) >> 16;
    return (unsigned short)r;
}

static __device__ __forceinline__ v8f vzero8() {
    v8f z = {0.f, 0.f, 0.f, 0.f, 0.f, 0.f, 0.f, 0.f};
    return z;
}

// ---------------- conversion kernels ----------------

__global__ void cvt_bf16(const float* __restrict__ in,
                         unsigned short* __restrict__ out, int n) {
    int i = blockIdx.x * blockDim.x + threadIdx.x;
    if (i < n) out[i] = f2bf(in[i]);
}

// in: [rows, cols] fp32 -> out: [cols, rows] bf16
__global__ void cvt_bf16_tr(const float* __restrict__ in,
                            unsigned short* __restrict__ out, int rows, int cols) {
    int i = blockIdx.x * blockDim.x + threadIdx.x;
    if (i < rows * cols) {
        int r = i / cols, c = i % cols;
        out[c * rows + r] = f2bf(in[i]);
    }
}

// ---------------- QKV projection GEMM ----------------
// A-fragment: 16x32 bf16, lane half interleave K {0-7,16-23}/{8-15,24-31}
static __device__ __forceinline__ void loadA(Frag* a, const unsigned short* __restrict__ X,
                                             int m0, int ln, int hl, int k) {
#pragma unroll
    for (int mt = 0; mt < 2; ++mt) {
        const unsigned short* p = X + (size_t)(m0 + mt * 16 + ln) * D_ + k + hl * 8;
        a[mt].q[0] = *(const uint4*)(p);
        a[mt].q[1] = *(const uint4*)(p + 16);
        __builtin_prefetch(p + 64, 0, 1);   // stream ahead -> global_prefetch_b8
    }
}

// B-fragment: 32x16 bf16, lane = column, lane half interleave K {0-7,16-23}/{8-15,24-31}
static __device__ __forceinline__ void loadB(Frag* b, const unsigned short* __restrict__ WT,
                                             int n0, int ln, int hl, int k) {
#pragma unroll
    for (int nt = 0; nt < 4; ++nt) {
        const unsigned short* p = WT + (size_t)(n0 + nt * 16 + ln) * D_ + k + hl * 8;
        b[nt].q[0] = *(const uint4*)(p);
        b[nt].q[1] = *(const uint4*)(p + 16);
    }
}

static __device__ __forceinline__ void mma8(v8f (&acc)[2][4], const Frag* a, const Frag* b) {
#pragma unroll
    for (int nt = 0; nt < 4; ++nt)
#pragma unroll
        for (int mt = 0; mt < 2; ++mt)
            acc[mt][nt] = __builtin_amdgcn_wmma_f32_16x16x32_bf16(
                false, a[mt].v, false, b[nt].v, (short)0, acc[mt][nt], false, false);
}

// X: [M=B*S, D] bf16.  WT: [U, D] bf16 (weight transposed).
// transposed==0: Out row-major [M,U] bf16 (Q, K)
// transposed==1: Out = V^T laid out [B][U][S] bf16
__global__ void __launch_bounds__(256)
gemm_qkv(const unsigned short* __restrict__ X,
         const unsigned short* __restrict__ WT,
         unsigned short* __restrict__ Out,
         int transposed) {
    const int lane = threadIdx.x & 31;
    const int wave = threadIdx.x >> 5;
    const int hl = lane >> 4;
    const int ln = lane & 15;

    const int m0 = blockIdx.x * 128 + (wave >> 1) * 32;   // 4 waves along M
    const int n0 = blockIdx.y * 128 + (wave & 1) * 64;    // 2 waves along N

    v8f acc[2][4];
#pragma unroll
    for (int mt = 0; mt < 2; ++mt)
#pragma unroll
        for (int nt = 0; nt < 4; ++nt) acc[mt][nt] = vzero8();

    // Software pipeline: double-buffered fragments, one K-step (32) lookahead.
    Frag a0[2], b0[4], a1[2], b1[4];
    loadA(a0, X, m0, ln, hl, 0);
    loadB(b0, WT, n0, ln, hl, 0);

    for (int k0 = 0; k0 < D_; k0 += 64) {
        loadA(a1, X, m0, ln, hl, k0 + 32);
        loadB(b1, WT, n0, ln, hl, k0 + 32);
        mma8(acc, a0, b0);                       // compute k0 while k0+32 in flight
        if (k0 + 64 < D_) {
            loadA(a0, X, m0, ln, hl, k0 + 64);
            loadB(b0, WT, n0, ln, hl, k0 + 64);
        }
        mma8(acc, a1, b1);                       // compute k0+32 while k0+64 in flight
    }

#pragma unroll
    for (int mt = 0; mt < 2; ++mt)
#pragma unroll
        for (int nt = 0; nt < 4; ++nt)
#pragma unroll
            for (int t = 0; t < 8; ++t) {
                int m = m0 + mt * 16 + t + 8 * hl;
                int n = n0 + nt * 16 + ln;
                unsigned short bfv = f2bf(acc[mt][nt][t]);
                if (!transposed) {
                    Out[(size_t)m * U_ + n] = bfv;
                } else {
                    int bb = m >> 11;          // m / S_
                    int ss = m & (S_ - 1);     // m % S_
                    Out[((size_t)bb * U_ + n) * S_ + ss] = bfv;
                }
            }
}

// ---------------- fused flash attention ----------------
// Q,K: [B*S, U] bf16 row-major.  Vt: [B, U, S] bf16.  Out: [B, S, U] fp32.
// One workgroup (8 waves) per 16-query tile. Wave w owns U-slice
// [w*128, w*128+128) for BOTH the score K-reduction and the P*V output.
__global__ void __launch_bounds__(256)
flash_attn(const unsigned short* __restrict__ Q,
           const unsigned short* __restrict__ K,
           const unsigned short* __restrict__ Vt,
           float* __restrict__ Out) {
    __shared__ float sc[16][32];   // score tile: 16 queries x 32 keys

    const int lane = threadIdx.x & 31;
    const int wave = threadIdx.x >> 5;
    const int hl = lane >> 4;
    const int ln = lane & 15;

    const int b  = blockIdx.x >> 7;            // S_/16 = 128 query tiles per batch
    const int q0 = (blockIdx.x & 127) * 16;

    const unsigned short* Qb = Q  + (size_t)b * S_ * U_;
    const unsigned short* Kb = K  + (size_t)b * S_ * U_;
    const unsigned short* Vb = Vt + (size_t)b * U_ * S_;

    const int us = wave * 128;                 // this wave's U slice

    // Preload Q fragments for this wave's K-slice of the score reduction.
    Frag qa[4];
#pragma unroll
    for (int kk = 0; kk < 4; ++kk) {
        const unsigned short* p = Qb + (size_t)(q0 + ln) * U_ + us + kk * 32 + hl * 8;
        qa[kk].q[0] = *(const uint4*)(p);
        qa[kk].q[1] = *(const uint4*)(p + 16);
    }

    v8f acc[8];
#pragma unroll
    for (int i = 0; i < 8; ++i) acc[i] = vzero8();

    float m_run = -1e30f;   // running row max (row = lane&15, replicated on lane pair)
    float l_run = 0.f;      // running row sum

    for (int kb = 0; kb < S_; kb += 32) {
        __syncthreads();
        for (int i = threadIdx.x; i < 16 * 32; i += 256) ((float*)sc)[i] = 0.f;
        __syncthreads();

        // Partial scores over this wave's 128-wide u slice.
        v8f s0 = vzero8(), s1 = vzero8();
#pragma unroll
        for (int kk = 0; kk < 4; ++kk) {
            Frag kb0, kb1;
            const unsigned short* p0 = Kb + (size_t)(kb + ln) * U_ + us + kk * 32 + hl * 8;
            kb0.q[0] = *(const uint4*)(p0);
            kb0.q[1] = *(const uint4*)(p0 + 16);
            const unsigned short* p1 = Kb + (size_t)(kb + 16 + ln) * U_ + us + kk * 32 + hl * 8;
            kb1.q[0] = *(const uint4*)(p1);
            kb1.q[1] = *(const uint4*)(p1 + 16);
            __builtin_prefetch(p0 + 32 * U_, 0, 1);
            s0 = __builtin_amdgcn_wmma_f32_16x16x32_bf16(
                false, qa[kk].v, false, kb0.v, (short)0, s0, false, false);
            s1 = __builtin_amdgcn_wmma_f32_16x16x32_bf16(
                false, qa[kk].v, false, kb1.v, (short)0, s1, false, false);
        }
        // Cross-wave reduce the K-slice partials in LDS.
#pragma unroll
        for (int t = 0; t < 8; ++t) {
            atomicAdd(&sc[t + 8 * hl][ln], s0[t]);
            atomicAdd(&sc[t + 8 * hl][ln + 16], s1[t]);
        }

        // Hoisted V-tile fragment loads: independent of the LDS score
        // reduction, so they overlap the barrier + softmax VALU phase.
        Frag vfr[8];
#pragma unroll
        for (int nt = 0; nt < 8; ++nt) {
            const unsigned short* p2 = Vb + (size_t)(us + nt * 16 + ln) * S_ + kb + hl * 8;
            vfr[nt].q[0] = *(const uint4*)(p2);
            vfr[nt].q[1] = *(const uint4*)(p2 + 16);
        }
        __syncthreads();

        // Online softmax. Lane handles row ln; its key columns are exactly the
        // A-fragment layout: base+0..7 and base+16..23, base = 8*hl.
        const int base = 8 * hl;
        float p[16];
        float mx = -1e30f;
#pragma unroll
        for (int i = 0; i < 8; ++i) { p[i]     = sc[ln][base + i];      mx = fmaxf(mx, p[i]); }
#pragma unroll
        for (int i = 0; i < 8; ++i) { p[8 + i] = sc[ln][base + 16 + i]; mx = fmaxf(mx, p[8 + i]); }
        mx = fmaxf(mx, __shfl_xor(mx, 16, 32));          // combine lane pair -> row max
        float m_new = fmaxf(m_run, mx);
        float sum = 0.f;
        Frag pf;
#pragma unroll
        for (int i = 0; i < 16; ++i) {
            float e = __expf(p[i] - m_new);
            sum += e;
            pf.h[i] = f2bf(e);
        }
        sum += __shfl_xor(sum, 16, 32);                  // row sum of new block
        float r = __expf(m_run - m_new);
        l_run = l_run * r + sum;
        m_run = m_new;

        // Rescale output accumulators: row of acc vgpr t is t + 8*hl; its
        // rescale factor lives at lane (t + 8*hl) -> ds_bpermute broadcast.
#pragma unroll
        for (int t = 0; t < 8; ++t) {
            float rv = __shfl(r, t + 8 * hl, 32);
#pragma unroll
            for (int nt = 0; nt < 8; ++nt) acc[nt][t] *= rv;
        }

        // P @ V over this wave's u output slice (fragments already resident).
#pragma unroll
        for (int nt = 0; nt < 8; ++nt)
            acc[nt] = __builtin_amdgcn_wmma_f32_16x16x32_bf16(
                false, pf.v, false, vfr[nt].v, (short)0, acc[nt], false, false);
    }

    // Final normalization and store (fp32 output).
    float linv = 1.f / l_run;
    float* Ob = Out + ((size_t)b * S_ + q0) * U_;
#pragma unroll
    for (int t = 0; t < 8; ++t) {
        float lv = __shfl(linv, t + 8 * hl, 32);
        int row = t + 8 * hl;
#pragma unroll
        for (int nt = 0; nt < 8; ++nt)
            Ob[(size_t)row * U_ + us + nt * 16 + ln] = acc[nt][t] * lv;
    }
}

// ---------------- host launcher ----------------

extern "C" void kernel_launch(void* const* d_in, const int* in_sizes, int n_in,
                              void* d_out, int out_size, void* d_ws, size_t ws_size,
                              hipStream_t stream) {
    const float* X  = (const float*)d_in[0];
    const float* Wq = (const float*)d_in[1];
    const float* Wk = (const float*)d_in[2];
    const float* Wv = (const float*)d_in[3];
    float* out = (float*)d_out;

    const size_t nX = (size_t)B_ * S_ * D_;   // 8M
    const size_t nW = (size_t)D_ * U_;        // 1M
    const size_t nQ = (size_t)B_ * S_ * U_;   // 8M

    unsigned short* Xb  = (unsigned short*)d_ws;
    unsigned short* WqT = Xb  + nX;
    unsigned short* WkT = WqT + nW;
    unsigned short* WvT = WkT + nW;
    unsigned short* Qb  = WvT + nW;
    unsigned short* Kb  = Qb  + nQ;
    unsigned short* Vt  = Kb  + nQ;           // total ~70 MB bf16 scratch

    cvt_bf16<<<(unsigned)((nX + 255) / 256), 256, 0, stream>>>(X, Xb, (int)nX);
    cvt_bf16_tr<<<(unsigned)((nW + 255) / 256), 256, 0, stream>>>(Wq, WqT, D_, U_);
    cvt_bf16_tr<<<(unsigned)((nW + 255) / 256), 256, 0, stream>>>(Wk, WkT, D_, U_);
    cvt_bf16_tr<<<(unsigned)((nW + 255) / 256), 256, 0, stream>>>(Wv, WvT, D_, U_);

    dim3 gg((B_ * S_) / 128, U_ / 128);
    gemm_qkv<<<gg, 256, 0, stream>>>(Xb, WqT, Qb, 0);
    gemm_qkv<<<gg, 256, 0, stream>>>(Xb, WkT, Kb, 0);
    gemm_qkv<<<gg, 256, 0, stream>>>(Xb, WvT, Vt, 1);

    flash_attn<<<B_ * (S_ / 16), 256, 0, stream>>>(Qb, Kb, Vt, out);
}
